// OptimizedPackedAttention_12300786335786
// MI455X (gfx1250) — matrix operand order repaired and checked by reading, not compiled, not run on hardware
//
#include <hip/hip_runtime.h>

#define T_TOK 2048
#define HID   1024
#define NH    16
#define DH    64
#define TP    2080   // padded Vt row length (allows b128 overreads past T)

typedef __attribute__((ext_vector_type(16))) _Float16 v16h;
typedef __attribute__((ext_vector_type(8)))  float    v8f;
typedef __attribute__((ext_vector_type(4)))  float    v4f;

union V16H { v16h v; v4f f[2]; };

// A-matrix fragment (16x32 f16), lane group g = lane>>4.
// Documented layout: lane m holds K pairs {8g..8g+7} then {16+8g..16+8g+7}.
__device__ __forceinline__ v16h load_frag_a(const _Float16* p, int g) {
  V16H u;
  u.f[0] = *reinterpret_cast<const v4f*>(p + 8 * g);
  u.f[1] = *reinterpret_cast<const v4f*>(p + 16 + 8 * g);
  return u.v;
}

// B-matrix fragment (32x16 f16): lane n holds column n, K = {8g..8g+7} then {16+8g..16+8g+7}.
__device__ __forceinline__ v16h load_frag_b(const _Float16* p, int g) {
  V16H u;
  u.f[0] = *reinterpret_cast<const v4f*>(p + 8 * g);
  u.f[1] = *reinterpret_cast<const v4f*>(p + 16 + 8 * g);
  return u.v;
}

__device__ __forceinline__ v8f wmma_f16(v16h a, v16h b, v8f c) {
  return __builtin_amdgcn_wmma_f32_16x16x32_f16(false, a, false, b, (short)0, c,
                                                false, false);
}

// ---------------------------------------------------------------- fp32 -> fp16
typedef __attribute__((ext_vector_type(4))) _Float16 v4h;
__global__ __launch_bounds__(256) void cvt16_kernel(const float* __restrict__ in,
                                                    _Float16* __restrict__ out,
                                                    int n4) {  // n/4 quads
  int i = blockIdx.x * blockDim.x + threadIdx.x;
  int stride = gridDim.x * blockDim.x;
  for (; i < n4; i += stride) {
    v4f x = reinterpret_cast<const v4f*>(in)[i];
    v4h h;
    h.x = (_Float16)x.x; h.y = (_Float16)x.y;
    h.z = (_Float16)x.z; h.w = (_Float16)x.w;
    reinterpret_cast<v4h*>(out)[i] = h;
  }
}

// -------------------------------------------------- GEMM  C = A * B^T + bias
// A: [M,K] f16 row-major, B: [N,K] f16 row-major (both contiguous along K).
// Each wave computes a 32x64 output tile (8 WMMA per K-step, distinct fragment
// registers so load-count waits can be partial). Block = 4 waves = 128x64.
// outF != nullptr -> fp32 row-major out. Else f16 out; transOut: [col*ldo+row].
__global__ __launch_bounds__(128) void gemm16_kernel(
    const _Float16* __restrict__ A, const _Float16* __restrict__ B,
    const float* __restrict__ bias, _Float16* __restrict__ outH,
    float* __restrict__ outF, int K, int N, int ldo, int transOut) {
  const int lane = threadIdx.x & 31;
  const int wave = threadIdx.x >> 5;
  const int g = lane >> 4;
  const int ln = lane & 15;
  const int mb = blockIdx.y * 128 + wave * 32;  // this wave's 32 output rows
  const int nb0 = blockIdx.x * 64;              // 64 output cols (4 frags)

  v8f acc[2][4] = {};
  const _Float16* arow0 = A + (size_t)(mb + ln) * K;
  const _Float16* arow1 = A + (size_t)(mb + 16 + ln) * K;
  const _Float16* br0 = B + (size_t)(nb0 + 0 * 16 + ln) * K;
  const _Float16* br1 = B + (size_t)(nb0 + 1 * 16 + ln) * K;
  const _Float16* br2 = B + (size_t)(nb0 + 2 * 16 + ln) * K;
  const _Float16* br3 = B + (size_t)(nb0 + 3 * 16 + ln) * K;

  for (int k0 = 0; k0 < K; k0 += 32) {
    __builtin_prefetch(arow0 + k0 + 256, 0, 3);
    const v16h a0 = load_frag_a(arow0 + k0, g);
    const v16h a1 = load_frag_a(arow1 + k0, g);
    const v16h b0 = load_frag_b(br0 + k0, g);
    const v16h b1 = load_frag_b(br1 + k0, g);
    const v16h b2 = load_frag_b(br2 + k0, g);
    const v16h b3 = load_frag_b(br3 + k0, g);
    acc[0][0] = wmma_f16(a0, b0, acc[0][0]);
    acc[0][1] = wmma_f16(a0, b1, acc[0][1]);
    acc[0][2] = wmma_f16(a0, b2, acc[0][2]);
    acc[0][3] = wmma_f16(a0, b3, acc[0][3]);
    acc[1][0] = wmma_f16(a1, b0, acc[1][0]);
    acc[1][1] = wmma_f16(a1, b1, acc[1][1]);
    acc[1][2] = wmma_f16(a1, b2, acc[1][2]);
    acc[1][3] = wmma_f16(a1, b3, acc[1][3]);
  }

#pragma unroll
  for (int mi = 0; mi < 2; ++mi) {
#pragma unroll
    for (int nb = 0; nb < 4; ++nb) {
      const int col = nb0 + nb * 16 + ln;
      const float bv = bias[col];
#pragma unroll
      for (int r = 0; r < 8; ++r) {
        const int row = mb + mi * 16 + r + 8 * g;  // C layout: VGPR r -> M=r+8g
        const float v = acc[mi][nb][r] + bv;
        if (outF)
          outF[(size_t)row * N + col] = v;
        else if (transOut)
          outH[(size_t)col * ldo + row] = (_Float16)v;
        else
          outH[(size_t)row * ldo + col] = (_Float16)v;
      }
    }
  }
}

// ------------------------------------------------- flash attention, 1 wave =
// one (16-row q tile, head); online softmax; block-diagonal segment mask.
__global__ __launch_bounds__(256) void attn16_kernel(
    const _Float16* __restrict__ Qh, const _Float16* __restrict__ Kh,
    const _Float16* __restrict__ Vt, const int* __restrict__ cu,
    _Float16* __restrict__ ctx) {
  __shared__ __align__(16) _Float16 pb[8][16][32];  // per-wave P scratch

  const int lane = threadIdx.x & 31;
  const int wave = threadIdx.x >> 5;
  const int g = lane >> 4;
  const int ln = lane & 15;
  const int unit = blockIdx.x * 8 + wave;
  const int head = unit & (NH - 1);
  const int t0 = (unit >> 4) * 16;

  int c9[9];
#pragma unroll
  for (int i = 0; i < 9; ++i) c9[i] = cu[i];
  auto sid = [&](int t) {
    int s = 0;
#pragma unroll
    for (int j = 1; j < 8; ++j) s += (t >= c9[j]);
    return s;
  };

  int rowseg[8];
#pragma unroll
  for (int r = 0; r < 8; ++r) rowseg[r] = sid(t0 + r + 8 * g);
  const int kstart = c9[sid(t0)];
  const int kend = c9[sid(t0 + 15) + 1];

  // Q fragments for this wave's 16 rows (dh=64 -> two K=32 chunks), kept live.
  const _Float16* qrow = Qh + (size_t)(t0 + ln) * HID + head * DH;
  const v16h qa0 = load_frag_a(qrow, g);
  const v16h qa1 = load_frag_a(qrow + 32, g);

  v8f o[4] = {};
  float mrow[8], lrow[8];
#pragma unroll
  for (int r = 0; r < 8; ++r) { mrow[r] = -1e30f; lrow[r] = 0.f; }
  const float scale = 0.125f;  // 1/sqrt(64)

  for (int kb = (kstart & ~31); kb < kend; kb += 32) {
    float pv[2][8];
#pragma unroll
    for (int hv = 0; hv < 2; ++hv) {  // two 16-key score tiles
      const int kt = kb + hv * 16 + ln;
      const int ktc = kt < (T_TOK - 1) ? kt : (T_TOK - 1);
      const _Float16* krow = Kh + (size_t)ktc * HID + head * DH;
      v8f sc = {};
      sc = wmma_f16(qa0, load_frag_b(krow, g), sc);
      sc = wmma_f16(qa1, load_frag_b(krow + 32, g), sc);
      const int kseg = (kt < kend) ? sid(kt) : -1;
#pragma unroll
      for (int r = 0; r < 8; ++r)
        pv[hv][r] = (kseg == rowseg[r]) ? sc[r] * scale : -1e30f;
    }
    // Online softmax over the 32 keys; row r+8g owned by this lane group.
#pragma unroll
    for (int r = 0; r < 8; ++r) {
      float mx = fmaxf(pv[0][r], pv[1][r]);
#pragma unroll
      for (int off = 8; off >= 1; off >>= 1)
        mx = fmaxf(mx, __shfl_xor(mx, off, 16));
      const float mnew = fmaxf(mrow[r], mx);
      const float corr = __expf(mrow[r] - mnew);
      const float p0 = pv[0][r] > -0.5e30f ? __expf(pv[0][r] - mnew) : 0.f;
      const float p1 = pv[1][r] > -0.5e30f ? __expf(pv[1][r] - mnew) : 0.f;
      float rs = p0 + p1;
#pragma unroll
      for (int off = 8; off >= 1; off >>= 1) rs += __shfl_xor(rs, off, 16);
      lrow[r] = lrow[r] * corr + rs;
      mrow[r] = mnew;
#pragma unroll
      for (int nb = 0; nb < 4; ++nb) o[nb][r] *= corr;
      pb[wave][r + 8 * g][ln] = (_Float16)p0;
      pb[wave][r + 8 * g][16 + ln] = (_Float16)p1;
    }
    asm volatile("" ::: "memory");  // same-wave LDS ops are in order (DScnt)
    // P (16x32) as A-fragment from LDS; V^T columns are contiguous b128 loads.
    const v16h pa = load_frag_a(&pb[wave][ln][0], g);
#pragma unroll
    for (int nb = 0; nb < 4; ++nb) {
      const _Float16* vp = Vt + (size_t)(head * DH + nb * 16 + ln) * TP + kb;
      o[nb] = wmma_f16(pa, load_frag_b(vp, g), o[nb]);
    }
    asm volatile("" ::: "memory");
  }

#pragma unroll
  for (int nb = 0; nb < 4; ++nb) {
#pragma unroll
    for (int r = 0; r < 8; ++r) {
      const float val = o[nb][r] / lrow[r];
      ctx[(size_t)(t0 + r + 8 * g) * HID + head * DH + nb * 16 + ln] =
          (_Float16)val;
    }
  }
}

extern "C" void kernel_launch(void* const* d_in, const int* in_sizes, int n_in,
                              void* d_out, int out_size, void* d_ws,
                              size_t ws_size, hipStream_t stream) {
  (void)in_sizes; (void)n_in; (void)out_size; (void)ws_size;
  const float* x  = (const float*)d_in[0];
  const int*   cu = (const int*)d_in[1];
  const float* Wq = (const float*)d_in[2];
  const float* bq = (const float*)d_in[3];
  const float* Wk = (const float*)d_in[4];
  const float* bk = (const float*)d_in[5];
  const float* Wv = (const float*)d_in[6];
  const float* bv = (const float*)d_in[7];
  const float* Wo = (const float*)d_in[8];
  const float* bo = (const float*)d_in[9];
  float* out = (float*)d_out;

  _Float16* p   = (_Float16*)d_ws;
  _Float16* xh  = p; p += (size_t)T_TOK * HID;
  _Float16* wqh = p; p += (size_t)HID * HID;
  _Float16* wkh = p; p += (size_t)HID * HID;
  _Float16* wvh = p; p += (size_t)HID * HID;
  _Float16* woh = p; p += (size_t)HID * HID;
  _Float16* Qh  = p; p += (size_t)T_TOK * HID;
  _Float16* Kh  = p; p += (size_t)T_TOK * HID;
  _Float16* Vth = p; p += (size_t)HID * TP;   // V^T, padded rows
  _Float16* Ch  = p; p += (size_t)T_TOK * HID;

  cvt16_kernel<<<256, 256, 0, stream>>>(x, xh, T_TOK * HID / 4);
  cvt16_kernel<<<256, 256, 0, stream>>>(Wq, wqh, HID * HID / 4);
  cvt16_kernel<<<256, 256, 0, stream>>>(Wk, wkh, HID * HID / 4);
  cvt16_kernel<<<256, 256, 0, stream>>>(Wv, wvh, HID * HID / 4);
  cvt16_kernel<<<256, 256, 0, stream>>>(Wo, woh, HID * HID / 4);

  dim3 gg(HID / 64, T_TOK / 128);
  gemm16_kernel<<<gg, 128, 0, stream>>>(xh, wqh, bq, Qh, nullptr, HID, HID, HID, 0);
  gemm16_kernel<<<gg, 128, 0, stream>>>(xh, wkh, bk, Kh, nullptr, HID, HID, HID, 0);
  gemm16_kernel<<<gg, 128, 0, stream>>>(xh, wvh, bv, Vth, nullptr, HID, HID, TP, 1);

  attn16_kernel<<<(T_TOK / 16 * NH) / 8, 256, 0, stream>>>(Qh, Kh, Vth, cu, Ch);

  gemm16_kernel<<<gg, 128, 0, stream>>>(Ch, woh, bo, nullptr, out, HID, HID, HID, 0);
}
